// EQGPT2Attention_70300024701864
// MI455X (gfx1250) — matrix operand order repaired and checked by reading, not compiled, not run on hardware
//
#include <hip/hip_runtime.h>

// ---------------------------------------------------------------------------
// EQGPT2 attention for MI455X (gfx1250, wave32).
//  * all GEMMs: v_wmma_f32_16x16x32_bf16, fp32 accumulation
//  * global->LDS staging via GLOBAL_LOAD_ASYNC_TO_LDS_B128 (ASYNCcnt),
//    double-buffered: tile t+1 streams in while tile t is computed
//  * operands pre-transposed in global memory so every staging is a straight
//    row copy (no register transpose on the hot path)
//  * causal structure: upper-triangle score blocks skipped, P@V K-loop trimmed
// ---------------------------------------------------------------------------

#define B_  2
#define S_  1024
#define D_  768
#define H_  12
#define EPSF 1e-5f
#define SCALEF 0.125f   // 1/sqrt(HEAD_DIM=64)
#define LDT 40          // LDS tile row stride (halfwords): 80B, 16B-aligned, conflict-free

typedef __attribute__((ext_vector_type(16))) __bf16 v16bf;
typedef __attribute__((ext_vector_type(8)))  float  v8f;

union ABFrag { v16bf v; unsigned int u[8]; };

__device__ __forceinline__ unsigned short f32_to_bf16(float f) {
  union { float f; unsigned int u; } c; c.f = f;
  unsigned int r = c.u + 0x7FFFu + ((c.u >> 16) & 1u);   // RNE
  return (unsigned short)(r >> 16);
}
__device__ __forceinline__ float bf16_to_f32(unsigned short h) {
  union { unsigned int u; float f; } c; c.u = ((unsigned int)h) << 16;
  return c.f;
}

#define WMMA_BF16(acc, a, b) \
  acc = __builtin_amdgcn_wmma_f32_16x16x32_bf16(false, (a), false, (b), (short)0, (acc), false, false)

// A fragment: 16x32 bf16 rows r0..r0+15; ISA striping (K-pairs per VGPR).
__device__ __forceinline__ v16bf load_frag_a(const unsigned short* sA, int r0, int lane) {
  ABFrag f;
  const unsigned short* p = sA + (r0 + (lane & 15)) * LDT;
  int kh = (lane >> 4) * 8;
#pragma unroll
  for (int i = 0; i < 8; ++i) {
    int k = ((i & 3) * 2) + ((i >> 2) * 16) + kh;
    f.u[i] = *(const unsigned int*)(p + k);
  }
  return f.v;
}
// B fragment: 32x16 from LDS tile holding B transposed as Bt[n][k].
__device__ __forceinline__ v16bf load_frag_b(const unsigned short* sB, int c0, int lane) {
  ABFrag f;
  const unsigned short* p = sB + (c0 + (lane & 15)) * LDT + (lane >> 4) * 8;
#pragma unroll
  for (int i = 0; i < 8; ++i) f.u[i] = *(const unsigned int*)(p + ((i & 3) * 2) + ((i >> 2) * 16));
  return f.v;
}

// ---- async double-buffered GEMM core --------------------------------------
// Block tile 128(M) x 128(N), K-step 32. 8 waves, each wave owns 32x64.
// sA/sB: 2 buffers of 128*LDT halfwords each. acc[8] = 2(M) x 4(N) frags.
template <int LDAG, int LDBG>
__device__ __forceinline__ void gemm_pipeline(const unsigned short* __restrict__ Ag,
                                              const unsigned short* __restrict__ Bg,
                                              int nk, unsigned short* sA, unsigned short* sB,
                                              int tid, int lane, int wm, int wn, v8f* acc) {
  unsigned aLds = (unsigned)(unsigned long long)(void*)sA;
  unsigned bLds = (unsigned)(unsigned long long)(void*)sB;
  // 512 b128 chunks per 128x32 tile -> 2 per thread; chunk: row=idx>>2, col=(idx&3)*8
  int r0 = tid >> 2,          c0 = (tid & 3) * 8;
  int r1 = (tid + 256) >> 2;  // c1 == c0

  auto issue = [&](int t) {
    int kt = t * 32;
    unsigned bo = (unsigned)((t & 1) * (128 * LDT * 2));   // byte offset of buffer
    unsigned lA0 = aLds + bo + (unsigned)(r0 * LDT + c0) * 2;
    unsigned lA1 = aLds + bo + (unsigned)(r1 * LDT + c0) * 2;
    unsigned lB0 = bLds + bo + (unsigned)(r0 * LDT + c0) * 2;
    unsigned lB1 = bLds + bo + (unsigned)(r1 * LDT + c0) * 2;
    unsigned long long gA0 = (unsigned long long)(size_t)(Ag + (size_t)r0 * LDAG + kt + c0);
    unsigned long long gA1 = (unsigned long long)(size_t)(Ag + (size_t)r1 * LDAG + kt + c0);
    unsigned long long gB0 = (unsigned long long)(size_t)(Bg + (size_t)r0 * LDBG + kt + c0);
    unsigned long long gB1 = (unsigned long long)(size_t)(Bg + (size_t)r1 * LDBG + kt + c0);
    asm volatile("global_load_async_to_lds_b128 %0, %4, off\n\t"
                 "global_load_async_to_lds_b128 %1, %5, off\n\t"
                 "global_load_async_to_lds_b128 %2, %6, off\n\t"
                 "global_load_async_to_lds_b128 %3, %7, off"
                 :: "v"(lA0), "v"(lA1), "v"(lB0), "v"(lB1),
                    "v"(gA0), "v"(gA1), "v"(gB0), "v"(gB1)
                 : "memory");
  };

  issue(0);
  for (int t = 0; t < nk; ++t) {
    if (t + 1 < nk) {
      issue(t + 1);                                        // prefetch next tile
      asm volatile("s_wait_asynccnt 0x4" ::: "memory");    // cur tile done (in-order)
    } else {
      asm volatile("s_wait_asynccnt 0x0" ::: "memory");
    }
    __syncthreads();
    const unsigned short* cA = sA + (t & 1) * (128 * LDT);
    const unsigned short* cB = sB + (t & 1) * (128 * LDT);
    v16bf fa0 = load_frag_a(cA, wm, lane);
    v16bf fa1 = load_frag_a(cA, wm + 16, lane);
    v16bf fb0 = load_frag_b(cB, wn, lane);
    v16bf fb1 = load_frag_b(cB, wn + 16, lane);
    v16bf fb2 = load_frag_b(cB, wn + 32, lane);
    v16bf fb3 = load_frag_b(cB, wn + 48, lane);
    WMMA_BF16(acc[0], fa0, fb0);
    WMMA_BF16(acc[1], fa0, fb1);
    WMMA_BF16(acc[2], fa0, fb2);
    WMMA_BF16(acc[3], fa0, fb3);
    WMMA_BF16(acc[4], fa1, fb0);
    WMMA_BF16(acc[5], fa1, fb1);
    WMMA_BF16(acc[6], fa1, fb2);
    WMMA_BF16(acc[7], fa1, fb3);
    __syncthreads();                                       // reads done before reuse of buffer
  }
}

// ---- Phase 0: LayerNorm (no mean-subtract) -> bf16 ------------------------
__global__ __launch_bounds__(256) void ln_kernel(const float* __restrict__ x,
                                                 unsigned short* __restrict__ hn) {
  int row = blockIdx.x;
  const float* xr = x + (size_t)row * D_;
  unsigned short* hr = hn + (size_t)row * D_;
  int tid = threadIdx.x;
  float v[3], s = 0.f, sq = 0.f;
#pragma unroll
  for (int i = 0; i < 3; ++i) {
    v[i] = xr[tid + i * 256];
    s += v[i]; sq += v[i] * v[i];
  }
#pragma unroll
  for (int off = 16; off > 0; off >>= 1) {
    s  += __shfl_down(s,  off, 32);
    sq += __shfl_down(sq, off, 32);
  }
  __shared__ float ls[8];
  __shared__ float lq[8];
  __shared__ float sinv;
  int lane = tid & 31, w = tid >> 5;
  if (lane == 0) { ls[w] = s; lq[w] = sq; }
  __syncthreads();
  if (tid == 0) {
    float ts = 0.f, tq = 0.f;
#pragma unroll
    for (int i = 0; i < 8; ++i) { ts += ls[i]; tq += lq[i]; }
    float mean = ts * (1.0f / D_);
    float var  = tq * (1.0f / D_) - mean * mean;
    sinv = rsqrtf(var + EPSF);
  }
  __syncthreads();
  float iv = sinv;
#pragma unroll
  for (int i = 0; i < 3; ++i) hr[tid + i * 256] = f32_to_bf16(v[i] * iv);
}

// ---- one-time weight transpose + f32->bf16:  Wt[h,n,d] = W[h,d,n] ---------
__global__ void cvtT_kernel(const float* __restrict__ in, unsigned short* __restrict__ out) {
  __shared__ unsigned short t[32][33];
  int h = blockIdx.z;
  int d0 = blockIdx.y * 32, n0 = blockIdx.x * 32;
  const float* src = in + ((size_t)h * D_ + d0) * D_ + n0;
  for (int r = threadIdx.y; r < 32; r += 8)
    t[r][threadIdx.x] = f32_to_bf16(src[(size_t)r * D_ + threadIdx.x]);
  __syncthreads();
  unsigned short* dst = out + ((size_t)h * D_ + n0) * D_ + d0;
  for (int r = threadIdx.y; r < 32; r += 8)
    dst[(size_t)r * D_ + threadIdx.x] = t[threadIdx.x][r];
}

// ---- self_score[b,h,t] = bqwkh[h,:] . hn[b,t,:] ---------------------------
__global__ __launch_bounds__(256) void self_kernel(const unsigned short* __restrict__ hn,
                                                   const float* __restrict__ bq,
                                                   float* __restrict__ self) {
  int idx = blockIdx.x * 8 + (threadIdx.x >> 5);
  int lane = threadIdx.x & 31;
  int b = idx / (H_ * S_);
  int h = (idx / S_) % H_;
  int t = idx % S_;
  const unsigned short* hr = hn + (size_t)(b * S_ + t) * D_;
  const float* bqr = bq + h * D_;
  float s = 0.f;
  for (int d = lane; d < D_; d += 32) s += bf16_to_f32(hr[d]) * bqr[d];
#pragma unroll
  for (int off = 16; off > 0; off >>= 1) s += __shfl_down(s, off, 32);
  if (lane == 0) self[idx] = s;
}

// ---- Phase 1: Out[b,h,s,n] = sum_d hn[b,s,d]*Wt[h,n,d]; bf16 out ----------
// out element (m,n) stored at plane + m*ldm + n*ldn  (Q: ldm=D,ldn=1 ; Vt: ldm=1,ldn=S)
__global__ __launch_bounds__(256) void gemm_hw_kernel(const unsigned short* __restrict__ hn,
                                                      const unsigned short* __restrict__ Wt,
                                                      unsigned short* __restrict__ out,
                                                      size_t ldm, size_t ldn) {
  __shared__ __align__(16) unsigned short sA[2 * 128 * LDT];
  __shared__ __align__(16) unsigned short sB[2 * 128 * LDT];
  int tid = threadIdx.x, lane = tid & 31, wv = tid >> 5;
  int b = blockIdx.z / H_, h = blockIdx.z % H_;
  int m0 = blockIdx.y * 128, n0 = blockIdx.x * 128;
  int wm = (wv & 3) * 32, wn = (wv >> 2) * 64;
  const unsigned short* Ag = hn + (size_t)(b * S_ + m0) * D_;
  const unsigned short* Bg = Wt + ((size_t)h * D_ + n0) * D_;
  v8f acc[8] = {};
  gemm_pipeline<D_, D_>(Ag, Bg, D_ / 32, sA, sB, tid, lane, wm, wn, acc);
  unsigned short* Og = out + (size_t)(b * H_ + h) * S_ * D_;
  int cn = lane & 15, hf = lane >> 4;
#pragma unroll
  for (int i = 0; i < 2; ++i)
#pragma unroll
    for (int j = 0; j < 4; ++j) {
#pragma unroll
      for (int r = 0; r < 8; ++r) {
        int m = m0 + wm + i * 16 + hf * 8 + r;
        int n = n0 + wn + j * 16 + cn;
        Og[(size_t)m * ldm + (size_t)n * ldn] = f32_to_bf16(acc[i * 4 + j][r]);
      }
    }
}

// ---- Phase 2: scores[b,h,i,t] = Q[b,h,i,:] . hn[b,t,:]; f32 out -----------
__global__ __launch_bounds__(256) void gemm_scores_kernel(const unsigned short* __restrict__ Q,
                                                          const unsigned short* __restrict__ hn,
                                                          float* __restrict__ scores) {
  int m0 = blockIdx.y * 128, n0 = blockIdx.x * 128;
  if (n0 >= m0 + 128) return;                 // fully above causal diagonal
  __shared__ __align__(16) unsigned short sA[2 * 128 * LDT];
  __shared__ __align__(16) unsigned short sB[2 * 128 * LDT];
  int tid = threadIdx.x, lane = tid & 31, wv = tid >> 5;
  int b = blockIdx.z / H_, h = blockIdx.z % H_;
  int wm = (wv & 3) * 32, wn = (wv >> 2) * 64;
  const unsigned short* Ag = Q + ((size_t)(b * H_ + h) * S_ + m0) * D_;
  const unsigned short* Bg = hn + (size_t)(b * S_ + n0) * D_;
  v8f acc[8] = {};
  gemm_pipeline<D_, D_>(Ag, Bg, D_ / 32, sA, sB, tid, lane, wm, wn, acc);
  float* Og = scores + ((size_t)(b * H_ + h) * S_ + m0) * S_ + n0;
  int cn = lane & 15, hf = lane >> 4;
#pragma unroll
  for (int i = 0; i < 2; ++i)
#pragma unroll
    for (int j = 0; j < 4; ++j) {
#pragma unroll
      for (int r = 0; r < 8; ++r) {
        int m = wm + i * 16 + hf * 8 + r;
        int n = wn + j * 16 + cn;
        Og[(size_t)m * S_ + n] = acc[i * 4 + j][r];
      }
    }
}

// ---- Phase 3: causal softmax; f32 probs -> d_out, bf16 copy -> P ----------
__global__ __launch_bounds__(256) void softmax_kernel(float* __restrict__ scores,
                                                      const float* __restrict__ self,
                                                      unsigned short* __restrict__ Pbf) {
  int row = blockIdx.x;                   // (b*H + h)*S + i
  int i = row % S_;
  int bh = row / S_;
  float* sr = scores + (size_t)row * S_;
  unsigned short* pr = Pbf + (size_t)row * S_;
  const float* selfr = self + (size_t)bh * S_;
  int tid = threadIdx.x, lane = tid & 31, w = tid >> 5;
  __shared__ float red[8];
  __shared__ float rbcast;
  float vals[4];
  float m = -3.4e38f;
#pragma unroll
  for (int t = 0; t < 4; ++t) {
    int j = tid + t * 256;
    float v = (j <= i) ? (sr[j] + selfr[j]) * SCALEF : -3.4e38f;
    vals[t] = v;
    m = fmaxf(m, v);
  }
#pragma unroll
  for (int off = 16; off > 0; off >>= 1) m = fmaxf(m, __shfl_xor(m, off, 32));
  if (lane == 0) red[w] = m;
  __syncthreads();
  if (tid == 0) {
    float mm = red[0];
#pragma unroll
    for (int k = 1; k < 8; ++k) mm = fmaxf(mm, red[k]);
    rbcast = mm;
  }
  __syncthreads();
  float M = rbcast;
  float s = 0.f;
#pragma unroll
  for (int t = 0; t < 4; ++t) {
    float e = __expf(vals[t] - M);
    vals[t] = e; s += e;
  }
#pragma unroll
  for (int off = 16; off > 0; off >>= 1) s += __shfl_xor(s, off, 32);
  if (lane == 0) red[w] = s;
  __syncthreads();
  if (tid == 0) {
    float ss = 0.f;
#pragma unroll
    for (int k = 0; k < 8; ++k) ss += red[k];
    rbcast = 1.0f / ss;
  }
  __syncthreads();
  float inv = rbcast;
#pragma unroll
  for (int t = 0; t < 4; ++t) {
    int j = tid + t * 256;
    float p = vals[t] * inv;
    sr[j] = p;
    pr[j] = f32_to_bf16(p);
  }
}

// ---- Phase 4: out[b,i,n] = sum_h P[b,h,i,:] . Vt[b,h,n,:] + bvo[n] --------
__global__ __launch_bounds__(256) void gemm_out_kernel(const unsigned short* __restrict__ P,
                                                       const unsigned short* __restrict__ Vt,
                                                       const float* __restrict__ bvo,
                                                       float* __restrict__ out) {
  __shared__ __align__(16) unsigned short sA[2 * 128 * LDT];
  __shared__ __align__(16) unsigned short sB[2 * 128 * LDT];
  int tid = threadIdx.x, lane = tid & 31, wv = tid >> 5;
  int b = blockIdx.z;
  int m0 = blockIdx.y * 128, n0 = blockIdx.x * 128;
  int wm = (wv & 3) * 32, wn = (wv >> 2) * 64;
  int nk = (m0 + 128) / 32;               // causal trim: P[i][j]==0 for j>i
  v8f acc[8] = {};
  for (int h = 0; h < H_; ++h) {
    const unsigned short* Ag = P + ((size_t)(b * H_ + h) * S_ + m0) * S_;
    const unsigned short* Bg = Vt + ((size_t)(b * H_ + h) * D_ + n0) * S_;
    gemm_pipeline<S_, S_>(Ag, Bg, nk, sA, sB, tid, lane, wm, wn, acc);
  }
  float* Og = out + ((size_t)(b * S_ + m0)) * D_ + n0;
  int cn = lane & 15, hf = lane >> 4;
#pragma unroll
  for (int i = 0; i < 2; ++i)
#pragma unroll
    for (int j = 0; j < 4; ++j) {
      float bv = bvo[n0 + wn + j * 16 + cn];
#pragma unroll
      for (int r = 0; r < 8; ++r) {
        int m = wm + i * 16 + hf * 8 + r;
        int n = wn + j * 16 + cn;
        Og[(size_t)m * D_ + n] = acc[i * 4 + j][r] + bv;
      }
    }
}

// ---------------------------------------------------------------------------
extern "C" void kernel_launch(void* const* d_in, const int* in_sizes, int n_in,
                              void* d_out, int out_size, void* d_ws, size_t ws_size,
                              hipStream_t stream) {
  const float* x     = (const float*)d_in[0];   // [B,S,D]
  const float* wqkh  = (const float*)d_in[1];   // [H,D,D]
  const float* bqwkh = (const float*)d_in[2];   // [H,D]
  const float* wvo   = (const float*)d_in[3];   // [H,D,D]
  const float* bvo   = (const float*)d_in[4];   // [D]

  float* out = (float*)d_out;                   // [B,S,D] then [B,H,S,S]
  float* aw  = out + (size_t)B_ * S_ * D_;

  // workspace (bf16 as u16), ~157 MB total, all 16B-aligned offsets
  unsigned short* hn  = (unsigned short*)d_ws;                 // B*S*D
  unsigned short* wqT = hn  + (size_t)B_ * S_ * D_;            // H*D*D (transposed)
  unsigned short* wvT = wqT + (size_t)H_ * D_ * D_;            // H*D*D (transposed)
  unsigned short* Qb  = wvT + (size_t)H_ * D_ * D_;            // B*H*S*D
  unsigned short* Vt  = Qb  + (size_t)B_ * H_ * S_ * D_;       // B*H*D*S (transposed)
  unsigned short* Pb  = Vt  + (size_t)B_ * H_ * S_ * D_;       // B*H*S*S
  float*          sb  = (float*)(Pb + (size_t)B_ * H_ * S_ * S_); // B*H*S

  ln_kernel<<<B_ * S_, 256, 0, stream>>>(x, hn);
  dim3 gT(D_ / 32, D_ / 32, H_), bT(32, 8);
  cvtT_kernel<<<gT, bT, 0, stream>>>(wqkh, wqT);
  cvtT_kernel<<<gT, bT, 0, stream>>>(wvo,  wvT);
  self_kernel<<<B_ * H_ * S_ / 8, 256, 0, stream>>>(hn, bqwkh, sb);

  dim3 gHW(D_ / 128, S_ / 128, B_ * H_);
  gemm_hw_kernel<<<gHW, 256, 0, stream>>>(hn, wqT, Qb, (size_t)D_, (size_t)1);  // Q[b,h,s,d]
  gemm_hw_kernel<<<gHW, 256, 0, stream>>>(hn, wvT, Vt, (size_t)1, (size_t)S_);  // Vt[b,h,n,s]

  dim3 gSC(S_ / 128, S_ / 128, B_ * H_);
  gemm_scores_kernel<<<gSC, 256, 0, stream>>>(Qb, hn, aw);

  softmax_kernel<<<B_ * H_ * S_, 256, 0, stream>>>(aw, sb, Pb);

  dim3 gOUT(D_ / 128, S_ / 128, B_);
  gemm_out_kernel<<<gOUT, 256, 0, stream>>>(Pb, Vt, bvo, out);
}
